// DCRNN_31482110279806
// MI455X (gfx1250) — compile-verified, matrix operand order checked by reading, not run on hardware
//
#include <hip/hip_runtime.h>
#include <hip/hip_bf16.h>
#include <stdint.h>

typedef __attribute__((ext_vector_type(16))) __bf16 v16bf;
typedef __attribute__((ext_vector_type(8)))  float  v8f;

#define NUM_NODES 10000
#define IN_FEAT   64
#define HID       256
#define OUTF      3
#define GATES     768   // 3*HID

// ---------------------------------------------------------------------------
// Conversion kernels
// ---------------------------------------------------------------------------
__global__ void cvt_f32_to_bf16(const float* __restrict__ in,
                                __bf16* __restrict__ out, int n) {
    int i = blockIdx.x * blockDim.x + threadIdx.x;
    if (i < n) out[i] = (__bf16)in[i];
}

// in: [K,N] row-major f32  ->  out: [N,K] row-major bf16 (i.e. B column-major)
__global__ void cvt_f32_to_bf16_T(const float* __restrict__ in,
                                  __bf16* __restrict__ out, int K, int N) {
    int i = blockIdx.x * blockDim.x + threadIdx.x;
    if (i >= K * N) return;
    int k = i / N, n = i % N;
    out[(size_t)n * K + k] = (__bf16)in[(size_t)k * N + n];
}

// ---------------------------------------------------------------------------
// Degree / normalization
// ---------------------------------------------------------------------------
__global__ void deg_init(float* __restrict__ deg, int n) {
    int i = blockIdx.x * blockDim.x + threadIdx.x;
    if (i < n) deg[i] = 1.0f;                        // self loop
}

__global__ void deg_edges(const int* __restrict__ dst, float* __restrict__ deg, int E) {
    int e = blockIdx.x * blockDim.x + threadIdx.x;
    if (e < E) atomicAdd(&deg[dst[e]], 1.0f);
}

__global__ void deg_rsqrt(const float* __restrict__ deg, float* __restrict__ dinv, int n) {
    int i = blockIdx.x * blockDim.x + threadIdx.x;
    if (i < n) dinv[i] = rsqrtf(deg[i]);
}

// ---------------------------------------------------------------------------
// WMMA GEMM: C[M,N] = A[M,K](bf16,row-major) * Bt[N,K](bf16, column of B contig)
// One 16x16 tile per wave; K stepped by 32 (v_wmma_f32_16x16x32_bf16).
// Per-lane operand layout per CDNA5 ISA 7.12.2 (16-bit A 16x32 table).
// ---------------------------------------------------------------------------
__global__ void __launch_bounds__(128)
wmma_gemm_bf16(const __bf16* __restrict__ A, const __bf16* __restrict__ Bt,
               float* __restrict__ C, int M, int N, int K) {
    int wave  = (int)((blockIdx.x * blockDim.x + threadIdx.x) >> 5);
    int lane  = threadIdx.x & 31;
    int tilesN = N >> 4;
    int tilesM = M >> 4;
    if (wave >= tilesM * tilesN) return;             // wave-uniform: EXEC stays full
    int tm = wave / tilesN;
    int tn = wave % tilesN;

    int l15 = lane & 15;
    int h   = lane >> 4;                             // half-wave selector
    const __bf16* arow = A  + (size_t)(tm * 16 + l15) * K;
    const __bf16* bcol = Bt + (size_t)(tn * 16 + l15) * K;

    v8f acc = {};
    for (int k = 0; k < K; k += 32) {
        v16bf a, b;
        int k0 = k + h * 8;
#pragma unroll
        for (int i = 0; i < 8; ++i) {
            a[i]     = arow[k0 + i];
            a[i + 8] = arow[k0 + 16 + i];
            b[i]     = bcol[k0 + i];
            b[i + 8] = bcol[k0 + 16 + i];
        }
        acc = __builtin_amdgcn_wmma_f32_16x16x32_bf16(
            false, a, false, b, (short)0, acc, false, false);
    }
    // C/D layout: lanes 0-15 rows 0..7, lanes 16-31 rows 8..15, col = lane&15
#pragma unroll
    for (int r = 0; r < 8; ++r) {
        int mm = tm * 16 + h * 8 + r;
        C[(size_t)mm * N + tn * 16 + l15] = acc[r];
    }
}

// ---------------------------------------------------------------------------
// GCN aggregation: agg = D^-1/2 (A+I) D^-1/2 xw
// ---------------------------------------------------------------------------
__global__ void gcn_selfloop(const float* __restrict__ xw, const float* __restrict__ dinv,
                             float* __restrict__ agg, int n) {
    int i = blockIdx.x * blockDim.x + threadIdx.x;   // over NUM_NODES*HID
    if (i >= n) return;
    int node = i >> 8;                               // /HID
    float s = dinv[node];
    agg[i] = xw[i] * s * s;
}

__global__ void gcn_scatter(const float* __restrict__ xw,
                            const int* __restrict__ src, const int* __restrict__ dst,
                            const float* __restrict__ dinv,
                            float* __restrict__ agg, int E) {
    long long gid = (long long)blockIdx.x * blockDim.x + threadIdx.x;
    int e = (int)(gid >> 6);                         // 64 lanes per edge
    if (e >= E) return;
    int f = ((int)gid & 63) << 2;                    // 4 floats per lane
    int s = src[e], d = dst[e];
    float norm = dinv[s] * dinv[d];
    const float4 v = *(const float4*)(xw + (size_t)s * HID + f);
    float* o = agg + (size_t)d * HID + f;
    atomicAdd(o + 0, norm * v.x);
    atomicAdd(o + 1, norm * v.y);
    atomicAdd(o + 2, norm * v.z);
    atomicAdd(o + 3, norm * v.w);
}

// bias + relu, emit bf16 activations for the next WMMA GEMM
__global__ void bias_relu_bf16(const float* __restrict__ agg, const float* __restrict__ bias,
                               __bf16* __restrict__ hb, int n) {
    int i = blockIdx.x * blockDim.x + threadIdx.x;
    if (i >= n) return;
    float v = agg[i] + bias[i & (HID - 1)];
    hb[i] = (__bf16)(v > 0.f ? v : 0.f);
}

// ---------------------------------------------------------------------------
// GRU sequential scan: single persistent workgroup (1024 thr = 32 waves).
// h and gate-vector live in LDS; W_hh streamed as bf16 from L2 each step;
// threads 768..799 prefetch next step's input-gate row.
// ---------------------------------------------------------------------------
__global__ void __launch_bounds__(1024)
gru_scan(const float* __restrict__ gi, const __bf16* __restrict__ Whh,
         const float* __restrict__ b_ih, const float* __restrict__ b_hh,
         float* __restrict__ hs, int T) {
    __shared__ float hcur[HID];
    __shared__ float gh[GATES];
    int t = threadIdx.x;
    if (t < HID) hcur[t] = 0.0f;
    __syncthreads();

    for (int step = 0; step < T; ++step) {
        if (t < GATES) {
            // gh[t] = dot(W_hh[t,:], hcur) + b_hh[t], bf16 weights unpacked
            const uint32_t* wrow = (const uint32_t*)(Whh + (size_t)t * HID);
            float acc = 0.0f;
#pragma unroll 8
            for (int kk = 0; kk < HID / 2; ++kk) {
                uint32_t p = wrow[kk];
                float w0 = __uint_as_float(p << 16);
                float w1 = __uint_as_float(p & 0xffff0000u);
                acc = fmaf(w0, hcur[2 * kk],     acc);
                acc = fmaf(w1, hcur[2 * kk + 1], acc);
            }
            gh[t] = acc + b_hh[t];
        } else if (step + 1 < T && t < GATES + 32) {
            // idle threads prefetch next step's gi row (128B strides)
            __builtin_prefetch(gi + (size_t)(step + 1) * GATES + (t - GATES) * 32, 0, 0);
        }
        __syncthreads();
        if (t < HID) {
            const float* g = gi + (size_t)step * GATES;
            float r  = 1.0f / (1.0f + __expf(-(g[t]           + b_ih[t]           + gh[t])));
            float z  = 1.0f / (1.0f + __expf(-(g[HID + t]     + b_ih[HID + t]     + gh[HID + t])));
            float nn = tanhf(g[2 * HID + t] + b_ih[2 * HID + t] + r * gh[2 * HID + t]);
            float hn = (1.0f - z) * nn + z * hcur[t];
            hs[(size_t)step * HID + t] = hn;
            hcur[t] = hn;
        }
        __syncthreads();
    }
}

// ---------------------------------------------------------------------------
// Output linear: out[n,o] = hs[n,:] . fc_w[o,:] + fc_b[o]
// ---------------------------------------------------------------------------
__global__ void fc_out_kernel(const float* __restrict__ hs, const float* __restrict__ fw,
                              const float* __restrict__ fb, float* __restrict__ out, int total) {
    int gid = blockIdx.x * blockDim.x + threadIdx.x;
    if (gid >= total) return;
    int n = gid / OUTF, o = gid % OUTF;
    const float* h = hs + (size_t)n * HID;
    const float* w = fw + (size_t)o * HID;
    float acc = fb[o];
#pragma unroll 8
    for (int k = 0; k < HID; ++k) acc = fmaf(h[k], w[k], acc);
    out[gid] = acc;
}

// ---------------------------------------------------------------------------
// Host-side launcher
// ---------------------------------------------------------------------------
extern "C" void kernel_launch(void* const* d_in, const int* in_sizes, int n_in,
                              void* d_out, int out_size, void* d_ws, size_t ws_size,
                              hipStream_t stream) {
    const float* x    = (const float*)d_in[0];
    const int*   ei   = (const int*)  d_in[1];
    const float* W1   = (const float*)d_in[2];
    const float* b1   = (const float*)d_in[3];
    const float* W2   = (const float*)d_in[4];
    const float* b2   = (const float*)d_in[5];
    const float* W_ih = (const float*)d_in[6];
    const float* W_hh = (const float*)d_in[7];
    const float* b_ih = (const float*)d_in[8];
    const float* b_hh = (const float*)d_in[9];
    const float* fc_w = (const float*)d_in[10];
    const float* fc_b = (const float*)d_in[11];
    float* out = (float*)d_out;

    const int E = in_sizes[1] / 2;
    const int* src = ei;
    const int* dst = ei + E;

    // ---- workspace carve-up (256B aligned) ----
    char* ws = (char*)d_ws;
    size_t off = 0;
    auto alloc = [&](size_t bytes) -> char* {
        char* p = ws + off;
        off = (off + bytes + 255) & ~(size_t)255;
        return p;
    };
    float*  deg  = (float*) alloc((size_t)NUM_NODES * 4);
    float*  dinv = (float*) alloc((size_t)NUM_NODES * 4);
    float*  xw   = (float*) alloc((size_t)NUM_NODES * HID * 4);
    float*  agg  = (float*) alloc((size_t)NUM_NODES * HID * 4);
    float*  gi   = (float*) alloc((size_t)NUM_NODES * GATES * 4);
    float*  hs   = (float*) alloc((size_t)NUM_NODES * HID * 4);
    __bf16* xb   = (__bf16*)alloc((size_t)NUM_NODES * HID * 2);  // activations (A)
    __bf16* w1t  = (__bf16*)alloc((size_t)HID * IN_FEAT * 2);    // [256,64]
    __bf16* w2t  = (__bf16*)alloc((size_t)HID * HID * 2);        // [256,256]
    __bf16* wih  = (__bf16*)alloc((size_t)GATES * HID * 2);      // [768,256] (already Bt)
    __bf16* whh  = (__bf16*)alloc((size_t)GATES * HID * 2);
    (void)ws_size; (void)n_in; (void)out_size;

    const int TPB = 256;
    auto blk = [](long long n, int t) { return (unsigned)((n + t - 1) / t); };

    // ---- one-time conversions ----
    cvt_f32_to_bf16<<<blk((long long)NUM_NODES * IN_FEAT, TPB), TPB, 0, stream>>>(
        x, xb, NUM_NODES * IN_FEAT);
    cvt_f32_to_bf16_T<<<blk((long long)IN_FEAT * HID, TPB), TPB, 0, stream>>>(
        W1, w1t, IN_FEAT, HID);
    cvt_f32_to_bf16_T<<<blk((long long)HID * HID, TPB), TPB, 0, stream>>>(
        W2, w2t, HID, HID);
    cvt_f32_to_bf16<<<blk((long long)GATES * HID, TPB), TPB, 0, stream>>>(
        W_ih, wih, GATES * HID);
    cvt_f32_to_bf16<<<blk((long long)GATES * HID, TPB), TPB, 0, stream>>>(
        W_hh, whh, GATES * HID);

    // ---- degree normalization ----
    deg_init<<<blk(NUM_NODES, TPB), TPB, 0, stream>>>(deg, NUM_NODES);
    deg_edges<<<blk(E, TPB), TPB, 0, stream>>>(dst, deg, E);
    deg_rsqrt<<<blk(NUM_NODES, TPB), TPB, 0, stream>>>(deg, dinv, NUM_NODES);

    const long long NH = (long long)NUM_NODES * HID;
    const unsigned gemm1_blocks = blk((long long)(NUM_NODES / 16) * (HID / 16), 4);   // 4 waves/blk
    const unsigned gemm2_blocks = gemm1_blocks;
    const unsigned gemm3_blocks = blk((long long)(NUM_NODES / 16) * (GATES / 16), 4);
    const unsigned scat_blocks  = blk((long long)E * 64, TPB);

    // ---- GCN layer 1 ----
    wmma_gemm_bf16<<<gemm1_blocks, 128, 0, stream>>>(xb, w1t, xw, NUM_NODES, HID, IN_FEAT);
    gcn_selfloop<<<blk(NH, TPB), TPB, 0, stream>>>(xw, dinv, agg, (int)NH);
    gcn_scatter<<<scat_blocks, TPB, 0, stream>>>(xw, src, dst, dinv, agg, E);
    bias_relu_bf16<<<blk(NH, TPB), TPB, 0, stream>>>(agg, b1, xb, (int)NH);

    // ---- GCN layer 2 ----
    wmma_gemm_bf16<<<gemm2_blocks, 128, 0, stream>>>(xb, w2t, xw, NUM_NODES, HID, HID);
    gcn_selfloop<<<blk(NH, TPB), TPB, 0, stream>>>(xw, dinv, agg, (int)NH);
    gcn_scatter<<<scat_blocks, TPB, 0, stream>>>(xw, src, dst, dinv, agg, E);
    bias_relu_bf16<<<blk(NH, TPB), TPB, 0, stream>>>(agg, b2, xb, (int)NH);

    // ---- GRU input gates: gi = h2 @ W_ih^T (W_ih rows are contiguous B columns) ----
    wmma_gemm_bf16<<<gemm3_blocks, 128, 0, stream>>>(xb, wih, gi, NUM_NODES, GATES, HID);

    // ---- sequential GRU scan (single persistent workgroup) ----
    gru_scan<<<1, 1024, 0, stream>>>(gi, whh, b_ih, b_hh, hs, NUM_NODES);

    // ---- output projection ----
    fc_out_kernel<<<blk((long long)NUM_NODES * OUTF, TPB), TPB, 0, stream>>>(
        hs, fc_w, fc_b, out, NUM_NODES * OUTF);
}
